// NodeModel_32169305047410
// MI455X (gfx1250) — matrix operand order repaired and checked by reading, not compiled, not run on hardware
//
#include <hip/hip_runtime.h>

typedef _Float16 v2h  __attribute__((ext_vector_type(2)));
typedef _Float16 v4h  __attribute__((ext_vector_type(4)));
typedef _Float16 v8h  __attribute__((ext_vector_type(8)));
typedef _Float16 v16h __attribute__((ext_vector_type(16)));
typedef float    v8f  __attribute__((ext_vector_type(8)));

#define N_NODES   100000
#define D_HID     256
#define D_OUT     64
#define K1PAD     160     // 144 padded to 5 k-tiles of 32
#define H_STRIDE  168     // halves; 336B rows -> bank-quad bijection for ds_load_b128
#define H1_STRIDE 264     // halves; 528B rows -> bank-quad bijection
#define MAX_DEG   64      // Poisson(12.5) in-degree; P(overflow) ~ 1e-28

// workspace layout
#define ELIST_BYTES ((size_t)N_NODES * MAX_DEG * 4)   // 25.6 MB
#define CNT_BYTES   ((size_t)N_NODES * 4)             // 400 KB
#define W1T_HALVES  (D_HID * K1PAD)                   // 256*160
#define W2T_HALVES  (D_OUT * D_HID)                   // 64*256

// ---------------- CSR-bucket build ----------------

__global__ void zero_cnt_kernel(int4* __restrict__ cnt4, int n4) {
    int i = blockIdx.x * blockDim.x + threadIdx.x;
    if (i < n4) cnt4[i] = make_int4(0, 0, 0, 0);
}

__global__ void build_elist_kernel(const int* __restrict__ eidx_row,
                                   int* __restrict__ cnt, int* __restrict__ elist, int E) {
    int e = blockIdx.x * blockDim.x + threadIdx.x;
    if (e >= E) return;
    int r = eidx_row[e];                       // edge_index[0][e]
    int p = atomicAdd(&cnt[r], 1);             // int atomic: native everywhere
    if (p < MAX_DEG) elist[r * MAX_DEG + p] = e;
}

// ---------------- weight prep: transpose + f16 convert + K-pad ----------------

__global__ void prep_weights_kernel(const float* __restrict__ W1, const float* __restrict__ W2,
                                    _Float16* __restrict__ w1t, _Float16* __restrict__ w2t) {
    int i = blockIdx.x * blockDim.x + threadIdx.x;
    if (i < W1T_HALVES) {
        int n = i / K1PAD, k = i % K1PAD;          // w1t[n][k] = W1[k][n]
        w1t[i] = (k < 144) ? (_Float16)W1[k * D_HID + n] : (_Float16)0.f;
    } else if (i < W1T_HALVES + W2T_HALVES) {
        int j = i - W1T_HALVES;
        int n = j / D_HID, k = j % D_HID;          // w2t[n][k] = W2[k][n]
        w2t[j] = (_Float16)W2[k * D_OUT + n];
    }
}

// ---------------- fused gather + MLP ----------------

// A fragment (16x32 f16, row-major source): lane<16 -> K {k0..k0+7, k0+16..k0+23},
// k0 = ktile*32 + (lane>>4)*8; M = lane&15.
__device__ __forceinline__ v16h load_a_frag(const _Float16* base, int stride, int lane, int ktile) {
    int m  = lane & 15;
    int k0 = ktile * 32 + ((lane >> 4) << 3);
    const _Float16* p = base + m * stride + k0;
    union { v16h v; v8h h[2]; } u;
    u.h[0] = *(const v8h*)(p);
    u.h[1] = *(const v8h*)(p + 16);
    return u.v;
}

// B fragment (32x16 f16) from K-contiguous transposed weights:
// lane: N = ntile*16 + (lane&15); K {kb..kb+7, kb+16..kb+23}, kb = ktile*32 + (lane>>4)*8.
__device__ __forceinline__ v16h load_b_frag(const _Float16* __restrict__ w, int stride,
                                            int lane, int ntile, int ktile) {
    int n  = ntile * 16 + (lane & 15);
    int kb = ktile * 32 + ((lane >> 4) << 3);
    const _Float16* p = w + n * stride + kb;
    union { v16h v; v8h h[2]; } u;
    u.h[0] = *(const v8h*)(p);
    u.h[1] = *(const v8h*)(p + 16);
    return u.v;
}

__global__ __launch_bounds__(128) void node_mlp_kernel(
        const float4* __restrict__ x4, const float2* __restrict__ ea2,
        const float4* __restrict__ u4, const int* __restrict__ batch,
        const int* __restrict__ cnt, const int* __restrict__ elist,
        const _Float16* __restrict__ w1t, const float* __restrict__ b1,
        const _Float16* __restrict__ w2t, const float* __restrict__ b2,
        float* __restrict__ out) {
    // 4 waves x 32 rows = 128 rows per block
    __shared__ __align__(16) _Float16 smem[128 * H_STRIDE + 128 * H1_STRIDE];
    _Float16* hbuf  = smem;                        // [128][H_STRIDE]
    _Float16* h1buf = smem + 128 * H_STRIDE;       // [128][H1_STRIDE]

    const int lane = threadIdx.x & 31;
    const int wave = threadIdx.x >> 5;             // each wave owns 32 rows (2 M-tiles)
    const int rowBase = blockIdx.x * 128 + wave * 32;

    _Float16* hw  = hbuf  + wave * 32 * H_STRIDE;
    _Float16* h1w = h1buf + wave * 32 * H1_STRIDE;

    // ---- phase 1: x cols [0,64), u[batch] cols [128,144), pad [144,160) ----
    for (int i = lane; i < 32 * 24; i += 32) {
        int r = i / 24, cg = i % 24;
        int g = rowBase + r;
        float vx = 0.f, vy = 0.f, vz = 0.f, vw = 0.f;
        int c4;
        if (cg < 16) {
            c4 = cg;
            if (g < N_NODES) { float4 t = x4[g * 16 + cg]; vx=t.x; vy=t.y; vz=t.z; vw=t.w; }
        } else {
            c4 = cg + 16;                          // 32..39 -> cols 128..159
            if (cg < 20 && g < N_NODES) {
                float4 t = u4[batch[g] * 4 + (cg - 16)]; vx=t.x; vy=t.y; vz=t.z; vw=t.w;
            }
        }
        v4h hv = { (_Float16)vx, (_Float16)vy, (_Float16)vz, (_Float16)vw };
        *(v4h*)(hw + r * H_STRIDE + c4 * 4) = hv;
    }

    // ---- phase 2: gather edge features, cols [64,128) ----
    // g is wave-uniform -> cnt/elist reads scalarize; each edge row is a
    // coalesced 256B load (lane handles 2 columns).
    for (int r = 0; r < 32; ++r) {
        int g = rowBase + r;
        float ax = 0.f, ay = 0.f;
        if (g < N_NODES) {
            int d = cnt[g];
            d = d > MAX_DEG ? MAX_DEG : d;
            const int* el = elist + g * MAX_DEG;
            for (int i = 0; i < d; ++i) {
                int e = el[i];
                float2 t = ea2[e * 32 + lane];
                ax += t.x; ay += t.y;
            }
        }
        v2h hv = { (_Float16)ax, (_Float16)ay };
        *(v2h*)(hw + r * H_STRIDE + 64 + lane * 2) = hv;
    }
    // Wave-private LDS tiles; same-wave DS ordering handled by dscnt waits.

    const int colL = lane & 15;
    const int rOff = (lane >> 4) << 3;

    // ---- GEMM1: 2 x [16 x 160] @ [160 x 256] -> ReLU -> h1 (f16, LDS) ----
    v16h a1[2][5];
#pragma unroll
    for (int mt = 0; mt < 2; ++mt)
#pragma unroll
        for (int kt = 0; kt < 5; ++kt)
            a1[mt][kt] = load_a_frag(hw + mt * 16 * H_STRIDE, H_STRIDE, lane, kt);

#pragma unroll
    for (int nt = 0; nt < 16; ++nt) {
        v16h bt[5];
#pragma unroll
        for (int kt = 0; kt < 5; ++kt) bt[kt] = load_b_frag(w1t, K1PAD, lane, nt, kt);
        int col = nt * 16 + colL;
        float bb = b1[col];
#pragma unroll
        for (int mt = 0; mt < 2; ++mt) {
            v8f acc = {};
#pragma unroll
            for (int kt = 0; kt < 5; ++kt)
                acc = __builtin_amdgcn_wmma_f32_16x16x32_f16(false, a1[mt][kt], false, bt[kt],
                                                             (short)0, acc, false, false);
#pragma unroll
            for (int r = 0; r < 8; ++r) {
                float v = acc[r] + bb;
                v = v > 0.f ? v : 0.f;
                h1w[(mt * 16 + r + rOff) * H1_STRIDE + col] = (_Float16)v;
            }
        }
    }

    // ---- GEMM2: 2 x [16 x 256] @ [256 x 64] + b2 -> out (f32) ----
#pragma unroll
    for (int mt = 0; mt < 2; ++mt) {
        v16h a2[8];
#pragma unroll
        for (int kt = 0; kt < 8; ++kt)
            a2[kt] = load_a_frag(h1w + mt * 16 * H1_STRIDE, H1_STRIDE, lane, kt);
#pragma unroll
        for (int nt = 0; nt < 4; ++nt) {
            v8f acc = {};
#pragma unroll
            for (int kt = 0; kt < 8; ++kt) {
                v16h b = load_b_frag(w2t, D_HID, lane, nt, kt);
                acc = __builtin_amdgcn_wmma_f32_16x16x32_f16(false, a2[kt], false, b,
                                                             (short)0, acc, false, false);
            }
            int col = nt * 16 + colL;
            float bb = b2[col];
#pragma unroll
            for (int r = 0; r < 8; ++r) {
                int g = rowBase + mt * 16 + r + rOff;
                if (g < N_NODES) out[g * 64 + col] = acc[r] + bb;
            }
        }
    }
}

// ---------------- launch ----------------

extern "C" void kernel_launch(void* const* d_in, const int* in_sizes, int n_in,
                              void* d_out, int out_size, void* d_ws, size_t ws_size,
                              hipStream_t stream) {
    const float* x    = (const float*)d_in[0];
    const int*   eidx = (const int*)  d_in[1];   // [2, E]; row = eidx[0..E)
    const float* ea   = (const float*)d_in[2];   // [E, 64]
    const float* u    = (const float*)d_in[3];   // [16, 16]
    const int*   bat  = (const int*)  d_in[4];   // [N]
    const float* W1   = (const float*)d_in[5];   // [144, 256]
    const float* b1   = (const float*)d_in[6];   // [256]
    const float* W2   = (const float*)d_in[7];   // [256, 64]
    const float* b2   = (const float*)d_in[8];   // [64]
    float* out = (float*)d_out;

    const int E = in_sizes[2] / 64;

    int*      elist = (int*)d_ws;
    int*      cnt   = (int*)((char*)d_ws + ELIST_BYTES);
    _Float16* w1t   = (_Float16*)((char*)d_ws + ELIST_BYTES + CNT_BYTES);
    _Float16* w2t   = w1t + W1T_HALVES;

    int n4 = N_NODES / 4;                         // 25000 int4
    zero_cnt_kernel<<<(n4 + 255) / 256, 256, 0, stream>>>((int4*)cnt, n4);

    build_elist_kernel<<<(E + 255) / 256, 256, 0, stream>>>(eidx, cnt, elist, E);

    int prepTot = W1T_HALVES + W2T_HALVES;
    prep_weights_kernel<<<(prepTot + 255) / 256, 256, 0, stream>>>(W1, W2, w1t, w2t);

    node_mlp_kernel<<<(N_NODES + 127) / 128, 128, 0, stream>>>(
        (const float4*)x, (const float2*)ea, (const float4*)u, bat,
        cnt, elist, w1t, b1, w2t, b2, out);
}
